// FlashHunyuanAttention_47356309405791
// MI455X (gfx1250) — matrix operand order repaired and checked by reading, not compiled, not run on hardware
//
#include <hip/hip_runtime.h>
#include <hip/hip_bf16.h>

// ---------------------------------------------------------------------------
// MI455X (gfx1250) fused attention block:
//   qkv = x @ Wqkv ; per-head RMSNorm+RoPE on q,k ; causal flash attention ;
//   out = attn @ Wo
// Matrix math via v_wmma_f32_16x16x32_bf16; LDS staging double-buffered and
// fed by GLOBAL_LOAD_ASYNC_TO_LDS_B128 + s_wait_asynccnt where available.
// ---------------------------------------------------------------------------

typedef __bf16 bf16_t;
typedef __attribute__((ext_vector_type(16))) __bf16 v16bf;
typedef __attribute__((ext_vector_type(8)))  __bf16 v8bf;
typedef __attribute__((ext_vector_type(4)))  __bf16 v4bf;
typedef __attribute__((ext_vector_type(2)))  __bf16 v2bf;
typedef __attribute__((ext_vector_type(8)))  float  v8f;
typedef __attribute__((ext_vector_type(4)))  int    v4i;

#define NB   2
#define SEQ  2048
#define HID  2048
#define NHD  16
#define HDD  128
#define NTOK (NB * SEQ)        // 4096
#define NQKV (3 * HID)         // 6144

// ---- async global->LDS path (gfx1250 ASYNCcnt), guarded -------------------
#if defined(__has_builtin)
#if __has_builtin(__builtin_amdgcn_global_load_async_to_lds_b128) && \
    __has_builtin(__builtin_amdgcn_s_wait_asynccnt)
#define HAVE_ASYNC_LDS 1
#endif
#endif
#ifndef HAVE_ASYNC_LDS
#define HAVE_ASYNC_LDS 0
#endif

static __device__ inline void async_copy16(const void* g, void* l) {
#if HAVE_ASYNC_LDS
  __builtin_amdgcn_global_load_async_to_lds_b128(
      (__attribute__((address_space(1))) v4i*)g,
      (__attribute__((address_space(3))) v4i*)l, 0, 0);
#else
  *(float4*)l = *(const float4*)g;
#endif
}
static __device__ inline void async_wait() {
#if HAVE_ASYNC_LDS
  __builtin_amdgcn_s_wait_asynccnt(0);
#endif
}

// ---- WMMA helpers (CDNA5 layouts, ISA 7.12.2) ------------------------------

static __device__ inline v16bf pack_a(v8bf lo, v8bf hi) {
  v16bf r;
#pragma unroll
  for (int i = 0; i < 8; ++i) { r[i] = lo[i]; r[i + 8] = hi[i]; }
  return r;
}

static __device__ inline v8f wmma_bf16(v16bf a, v16bf b, v8f c) {
  return __builtin_amdgcn_wmma_f32_16x16x32_bf16(false, a, false, b,
                                                 (short)0, c, false, false);
}

// ---- 1) f32 -> bf16 conversion --------------------------------------------

__global__ void cvt_bf16_kernel(const float* __restrict__ in,
                                bf16_t* __restrict__ out, int n4) {
  int i = blockIdx.x * blockDim.x + threadIdx.x;
  if (i >= n4) return;
  const float4 v = ((const float4*)in)[i];
  v4bf o;
  o[0] = (bf16_t)v.x; o[1] = (bf16_t)v.y; o[2] = (bf16_t)v.z; o[3] = (bf16_t)v.w;
  ((v4bf*)out)[i] = o;
}

// ---- 2/5) Pipelined WMMA GEMM: C[M,N] = A[M,K] * B[K,N] --------------------
// Block tile 128x128, 256 threads = 8 waves (4M x 2N), each wave 2x4 WMMA
// tiles, K step 32, double-buffered LDS, async A staging, b32-packed B
// transpose staging (K contiguous in LDS so every B fragment is one 32B load).

template <bool OUT_BF16>
__global__ __launch_bounds__(256)
void gemm_wmma_kernel(const bf16_t* __restrict__ A, const bf16_t* __restrict__ Bm,
                      void* __restrict__ Cout, int M, int N, int K) {
  __shared__ bf16_t As[2][128 * 32];   // [row][k]
  __shared__ bf16_t Bst[2][128 * 32];  // [n][k], k contiguous

  const int tid  = threadIdx.x;
  const int lane = tid & 31;
  const int wave = tid >> 5;
  const int wm   = wave & 3;
  const int wn   = wave >> 2;
  const int bm   = blockIdx.y * 128;
  const int bn   = blockIdx.x * 128;

  v8f acc[2][4];
#pragma unroll
  for (int mi = 0; mi < 2; ++mi)
#pragma unroll
    for (int ni = 0; ni < 4; ++ni)
#pragma unroll
      for (int r = 0; r < 8; ++r) acc[mi][ni][r] = 0.0f;

  // staging roles
  const int a_row = tid >> 1;          // 0..127
  const int a_col = (tid & 1) * 16;    // 0/16
  const int b_kp  = (tid & 15) * 2;    // k pair 0,2,..,30
  const int b_n0  = (tid >> 4) * 8;    // 0..120

  const int l16  = lane & 15;
  const int kh8  = (lane >> 4) * 8;
  const int kg16 = (lane >> 4) * 16;

  auto stage_a_async = [&](int kt, int pb) {
    const bf16_t* ag = A + (size_t)(bm + a_row) * K + kt + a_col;
    bf16_t* dst = &As[pb][a_row * 32 + a_col];
    async_copy16(ag, dst);
    async_copy16(ag + 8, dst + 8);
  };
  auto load_b_regs = [&](int kt, v8bf br[2]) {
    const bf16_t* bg = Bm + (size_t)(kt + b_kp) * N + bn + b_n0;
    br[0] = *(const v8bf*)(bg);
    br[1] = *(const v8bf*)(bg + N);
  };
  auto scatter_b = [&](const v8bf br[2], int pb) {
#pragma unroll
    for (int i = 0; i < 8; ++i) {
      v2bf pr; pr[0] = br[0][i]; pr[1] = br[1][i];
      *(v2bf*)&Bst[pb][(b_n0 + i) * 32 + b_kp] = pr;   // one b32 store
    }
  };

  // prologue: tile 0
  {
    v8bf b0[2];
    stage_a_async(0, 0);
    load_b_regs(0, b0);
    scatter_b(b0, 0);
    async_wait();
    __syncthreads();
  }

  const int niter = K / 32;
  for (int it = 0; it < niter; ++it) {
    const int p = it & 1;
    const bool more = (it + 1) < niter;
    v8bf bnext[2];
    if (more) {
      stage_a_async((it + 1) * 32, p ^ 1);   // async -> other buffer
      load_b_regs((it + 1) * 32, bnext);     // global latency hidden by wmma
    }

    v16bf af[2];
#pragma unroll
    for (int mi = 0; mi < 2; ++mi) {
      const bf16_t* ap = &As[p][(wm * 32 + mi * 16 + l16) * 32];
      af[mi] = pack_a(*(const v8bf*)(ap + kg16), *(const v8bf*)(ap + kg16 + 8));
    }
    v16bf bf[4];
#pragma unroll
    for (int ni = 0; ni < 4; ++ni)
      bf[ni] = *(const v16bf*)&Bst[p][(wn * 64 + ni * 16 + l16) * 32 + kg16];

#pragma unroll
    for (int mi = 0; mi < 2; ++mi)
#pragma unroll
      for (int ni = 0; ni < 4; ++ni)
        acc[mi][ni] = wmma_bf16(af[mi], bf[ni], acc[mi][ni]);

    if (more) scatter_b(bnext, p ^ 1);
    async_wait();
    __syncthreads();
  }

  const int rh = (lane >> 4) * 8;
#pragma unroll
  for (int mi = 0; mi < 2; ++mi)
#pragma unroll
    for (int ni = 0; ni < 4; ++ni)
#pragma unroll
      for (int r = 0; r < 8; ++r) {
        const int row = bm + wm * 32 + mi * 16 + rh + r;
        const int col = bn + wn * 64 + ni * 16 + l16;
        const float v = acc[mi][ni][r];
        if (OUT_BF16) ((bf16_t*)Cout)[(size_t)row * N + col] = (bf16_t)v;
        else          ((float*)Cout)[(size_t)row * N + col]  = v;
      }
}

// ---- 3) fused per-head RMSNorm + RoPE on q,k (in place, f32 math) ----------

static __device__ inline float block_sum_128(float v, float* red) {
  const int t = threadIdx.x;
#pragma unroll
  for (int m = 16; m >= 1; m >>= 1) v += __shfl_xor(v, m, 32);
  if ((t & 31) == 0) red[t >> 5] = v;
  __syncthreads();
  const float s = red[0] + red[1] + red[2] + red[3];
  __syncthreads();
  return s;
}

__global__ __launch_bounds__(128)
void normrope_kernel(bf16_t* __restrict__ qkv, const float* __restrict__ qw,
                     const float* __restrict__ kw) {
  __shared__ float red[4];
  __shared__ float qsh[128];
  __shared__ float ksh[128];
  const int i    = threadIdx.x;
  const int head = blockIdx.x % NHD;
  const int tok  = blockIdx.x / NHD;
  const int pos  = tok % SEQ;

  bf16_t* qp = qkv + (size_t)tok * NQKV + head * HDD;
  bf16_t* kp = qp + HID;
  const float q = (float)qp[i];
  const float k = (float)kp[i];

  const float sq = block_sum_128(q * q, red);
  const float sk = block_sum_128(k * k, red);
  const float qn = q * rsqrtf(sq * (1.0f / 128.0f) + 1e-6f) * qw[i];
  const float kn = k * rsqrtf(sk * (1.0f / 128.0f) + 1e-6f) * kw[i];
  qsh[i] = qn; ksh[i] = kn;
  __syncthreads();

  const int   j   = i & 63;
  const float fr  = (float)pos * __powf(10000.0f, -(float)(2 * j) * (1.0f / 128.0f));
  const float c   = __cosf(fr);
  const float s   = __sinf(fr);
  const float qro = (i < 64) ? (qn * c - qsh[i + 64] * s) : (qn * c + qsh[i - 64] * s);
  const float kro = (i < 64) ? (kn * c - ksh[i + 64] * s) : (kn * c + ksh[i - 64] * s);
  qp[i] = (bf16_t)qro;
  kp[i] = (bf16_t)kro;
}

// ---- 4) causal flash attention ---------------------------------------------
// 128 threads = 4 waves x 16 q rows. Key blocks of 32, double-buffered K/V
// staging (K async row-major, V b32-packed transpose). Row sums computed by an
// extra WMMA against an all-ones B fragment (no ds_bpermute for sums).

__global__ __launch_bounds__(128)
void flash_attn_kernel(const bf16_t* __restrict__ qkv, bf16_t* __restrict__ attn) {
  __shared__ bf16_t Ks[2][32 * 128];    // [key][d]
  __shared__ bf16_t Vt[2][128 * 32];    // [d][key], key contiguous
  __shared__ bf16_t Ps[4][16 * 32];     // per-wave P relayout scratch

  const int tid  = threadIdx.x;
  const int lane = tid & 31;
  const int wave = tid >> 5;
  const int qt   = blockIdx.x % (SEQ / 64);
  const int h    = (blockIdx.x / (SEQ / 64)) % NHD;
  const int b    = blockIdx.x / ((SEQ / 64) * NHD);
  const int qbase = qt * 64 + wave * 16;
  const size_t bS = (size_t)b * SEQ;
  const int hoff  = h * HDD;

  const float scale = 0.08838834764831845f;  // 1/sqrt(128)

  const int l16  = lane & 15;
  const int kh8  = (lane >> 4) * 8;
  const int kg16 = (lane >> 4) * 16;
  const int rh   = (lane >> 4) * 8;

  // Q A-fragments, 4 chunks of K=32 over head dim
  v16bf qf[4];
  {
    const bf16_t* qg = qkv + (bS + qbase + l16) * NQKV + hoff;
#pragma unroll
    for (int c = 0; c < 4; ++c)
      qf[c] = pack_a(*(const v8bf*)(qg + c * 32 + kg16),
                     *(const v8bf*)(qg + c * 32 + kg16 + 8));
  }

  v16bf ones;
#pragma unroll
  for (int i = 0; i < 16; ++i) ones[i] = (bf16_t)1.0f;

  v8f acc[8], accl;
  float mrow[8];
#pragma unroll
  for (int n = 0; n < 8; ++n)
#pragma unroll
    for (int r = 0; r < 8; ++r) acc[n][r] = 0.0f;
#pragma unroll
  for (int r = 0; r < 8; ++r) { accl[r] = 0.0f; mrow[r] = -1e30f; }

  // staging roles
  const int k_key = tid >> 2;          // 0..31
  const int k_d0  = (tid & 3) * 32;    // 4x16B async
  const int v_kp  = (tid & 15) * 2;    // key pair
  const int v_d0  = (tid >> 4) * 16;   // 16 d values

  auto stage_k_async = [&](int k0, int pb) {
    const bf16_t* kg = qkv + (bS + k0 + k_key) * NQKV + hoff + HID + k_d0;
    bf16_t* dst = &Ks[pb][k_key * 128 + k_d0];
#pragma unroll
    for (int c = 0; c < 4; ++c) async_copy16(kg + c * 8, dst + c * 8);
  };
  auto load_v_regs = [&](int k0, v8bf vr[4]) {
    const bf16_t* vg = qkv + (bS + k0 + v_kp) * NQKV + hoff + 2 * HID + v_d0;
    vr[0] = *(const v8bf*)(vg);
    vr[1] = *(const v8bf*)(vg + 8);
    vr[2] = *(const v8bf*)(vg + NQKV);
    vr[3] = *(const v8bf*)(vg + NQKV + 8);
  };
  auto scatter_v = [&](const v8bf vr[4], int pb) {
#pragma unroll
    for (int i = 0; i < 8; ++i) {
      v2bf p0; p0[0] = vr[0][i]; p0[1] = vr[2][i];
      *(v2bf*)&Vt[pb][(v_d0 + i) * 32 + v_kp] = p0;
      v2bf p1; p1[0] = vr[1][i]; p1[1] = vr[3][i];
      *(v2bf*)&Vt[pb][(v_d0 + 8 + i) * 32 + v_kp] = p1;
    }
  };

  const int nkb = qt * 2 + 2;   // causal coverage of rows <= qt*64+63

  {
    v8bf v0[4];
    stage_k_async(0, 0);
    load_v_regs(0, v0);
    scatter_v(v0, 0);
    async_wait();
    __syncthreads();
  }

  for (int kb = 0; kb < nkb; ++kb) {
    const int p = kb & 1;
    const int k0 = kb * 32;
    const bool more = (kb + 1) < nkb;
    v8bf vnext[4];
    if (more) {
      stage_k_async(k0 + 32, p ^ 1);
      load_v_regs(k0 + 32, vnext);
    }

    // scores: 2 sub-tiles of 16 keys over 4 head-dim chunks
    v8f sc[2];
#pragma unroll
    for (int jt = 0; jt < 2; ++jt) {
#pragma unroll
      for (int r = 0; r < 8; ++r) sc[jt][r] = 0.0f;
#pragma unroll
      for (int c = 0; c < 4; ++c) {
        v16bf kf = *(const v16bf*)&Ks[p][(jt * 16 + l16) * 128 + c * 32 + kg16];
        sc[jt] = wmma_bf16(qf[c], kf, sc[jt]);
      }
    }

    // online softmax: mask, row max (shfl), exp; sums via WMMA below
#pragma unroll
    for (int r = 0; r < 8; ++r) {
      const int qglob = qbase + rh + r;
      float s0 = sc[0][r] * scale;
      float s1 = sc[1][r] * scale;
      s0 = (k0 + l16      <= qglob) ? s0 : -1e30f;
      s1 = (k0 + 16 + l16 <= qglob) ? s1 : -1e30f;
      float vmax = fmaxf(s0, s1);
#pragma unroll
      for (int m = 8; m >= 1; m >>= 1) vmax = fmaxf(vmax, __shfl_xor(vmax, m, 32));
      const float mnew  = fmaxf(mrow[r], vmax);
      const float alpha = __expf(mrow[r] - mnew);
      mrow[r] = mnew;
      sc[0][r] = __expf(s0 - mnew);
      sc[1][r] = __expf(s1 - mnew);
      accl[r] = accl[r] * alpha;
#pragma unroll
      for (int n = 0; n < 8; ++n) acc[n][r] = acc[n][r] * alpha;
    }

    // P: C-layout -> bf16 -> LDS -> A-fragment (intra-wave, DS-ordered)
    bf16_t* ps = &Ps[wave][0];
#pragma unroll
    for (int jt = 0; jt < 2; ++jt)
#pragma unroll
      for (int r = 0; r < 8; ++r)
        ps[(rh + r) * 32 + jt * 16 + l16] = (bf16_t)sc[jt][r];
    v16bf pf = pack_a(*(const v8bf*)(ps + l16 * 32 + kg16),
                      *(const v8bf*)(ps + l16 * 32 + kg16 + 8));

    // l += rowsum(P) via matrix unit; attn += P @ V
    accl = wmma_bf16(pf, ones, accl);
#pragma unroll
    for (int n = 0; n < 8; ++n) {
      v16bf vf = *(const v16bf*)&Vt[p][(n * 16 + l16) * 32 + kg16];
      acc[n] = wmma_bf16(pf, vf, acc[n]);
    }

    if (more) scatter_v(vnext, p ^ 1);
    async_wait();
    __syncthreads();
  }

  // normalize and write attn output [B,S,nh,hd] (bf16)
  float inv[8];
#pragma unroll
  for (int r = 0; r < 8; ++r) inv[r] = 1.0f / accl[r];
#pragma unroll
  for (int n = 0; n < 8; ++n)
#pragma unroll
    for (int r = 0; r < 8; ++r) {
      const int row = qbase + rh + r;
      attn[(bS + row) * HID + hoff + n * 16 + l16] = (bf16_t)(acc[n][r] * inv[r]);
    }
}

// ---- launch ----------------------------------------------------------------

extern "C" void kernel_launch(void* const* d_in, const int* in_sizes, int n_in,
                              void* d_out, int out_size, void* d_ws, size_t ws_size,
                              hipStream_t stream) {
  const float* x        = (const float*)d_in[0];
  const float* w_qkv    = (const float*)d_in[1];
  const float* w_o      = (const float*)d_in[2];
  const float* q_norm_w = (const float*)d_in[3];
  const float* k_norm_w = (const float*)d_in[4];

  char* ws = (char*)d_ws;
  bf16_t* xb    = (bf16_t*)ws;  ws += (size_t)NTOK * HID  * 2;
  bf16_t* wqb   = (bf16_t*)ws;  ws += (size_t)HID  * NQKV * 2;
  bf16_t* wob   = (bf16_t*)ws;  ws += (size_t)HID  * HID  * 2;
  bf16_t* qkvb  = (bf16_t*)ws;  ws += (size_t)NTOK * NQKV * 2;
  bf16_t* attnb = (bf16_t*)ws;  ws += (size_t)NTOK * HID  * 2;

  {
    const int nx = NTOK * HID / 4, nq = HID * NQKV / 4, no = HID * HID / 4;
    cvt_bf16_kernel<<<(nx + 255) / 256, 256, 0, stream>>>(x, xb, nx);
    cvt_bf16_kernel<<<(nq + 255) / 256, 256, 0, stream>>>(w_qkv, wqb, nq);
    cvt_bf16_kernel<<<(no + 255) / 256, 256, 0, stream>>>(w_o, wob, no);
  }
  gemm_wmma_kernel<true><<<dim3(NQKV / 128, NTOK / 128), 256, 0, stream>>>(
      xb, wqb, qkvb, NTOK, NQKV, HID);
  normrope_kernel<<<NTOK * NHD, 128, 0, stream>>>(qkvb, q_norm_w, k_norm_w);
  flash_attn_kernel<<<NB * NHD * (SEQ / 64), 128, 0, stream>>>(qkvb, attnb);
  gemm_wmma_kernel<false><<<dim3(HID / 128, NTOK / 128), 256, 0, stream>>>(
      attnb, wob, d_out, NTOK, HID, HID);
}
